// GraphUNet_54511724920929
// MI455X (gfx1250) — matrix operand order repaired and checked by reading, not compiled, not run on hardware
//
#include <hip/hip_runtime.h>
#include <hip/hip_bf16.h>

typedef __bf16 bf16_t;
typedef __attribute__((ext_vector_type(16))) __bf16 v16bf;
typedef __attribute__((ext_vector_type(8)))  float  v8f;
typedef __attribute__((ext_vector_type(4)))  unsigned int u32x4;
typedef __attribute__((ext_vector_type(8)))  int i32x8;
typedef __attribute__((ext_vector_type(4)))  int i32x4;
typedef __attribute__((ext_vector_type(4)))  int v4i;

#define CH 128

// ---- CDNA5 async data-movement availability (device pass only) -------------
#if defined(__has_builtin)
#if __has_builtin(__builtin_amdgcn_global_load_async_to_lds_b128) && \
    __has_builtin(__builtin_amdgcn_s_wait_asynccnt)
#define HAVE_ASYNC_LDS 1
#endif
#if __has_builtin(__builtin_amdgcn_tensor_load_to_lds) && \
    __has_builtin(__builtin_amdgcn_s_wait_tensorcnt)
#define HAVE_TDM 1
#endif
#endif

// AS1 (global) / AS3 (LDS) v4i pointers as required by the async-LDS builtin.
#define GLOBAL_V4I(p) ((__attribute__((address_space(1))) v4i*)(p))
#define LDS_V4I(p)    ((__attribute__((address_space(3))) v4i*)(p))

// ---------------------------------------------------------------------------
// bf16 WMMA GEMM: C (f32, n x n) = A (bf16, n x n) @ B (bf16, n x n), n % 64 == 0
// Block = 128 threads (4 waves). Block tile 64x64, each wave a 32x32 subtile
// built from 2x2 V_WMMA_F32_16X16X32_BF16 ops per 32-wide K step.
// A tile staged via TDM (tensor_load_to_lds) when available; B tile via
// GLOBAL_LOAD_ASYNC_TO_LDS_B128 when available; uint4 staging otherwise.
// ---------------------------------------------------------------------------
__global__ __launch_bounds__(128) void gemm_bf16_sq(const bf16_t* A,
                                                    const bf16_t* B,
                                                    float* __restrict__ C,
                                                    int n) {
  __shared__ __align__(16) bf16_t sA[64][32];   // M x K tile
  __shared__ __align__(16) bf16_t sB[32][64];   // K x N tile

  const int tid  = threadIdx.x;
  const int wave = tid >> 5;
  const int lane = tid & 31;
  const int l16  = lane & 15;
  const int hi   = lane >> 4;          // 0: lanes 0-15, 1: lanes 16-31
  const int m0   = blockIdx.y * 64;
  const int n0   = blockIdx.x * 64;
  const int wm   = (wave >> 1) * 32;   // wave row offset inside block tile
  const int wn   = (wave & 1) * 32;    // wave col offset inside block tile

  v8f acc[2][2];
#pragma unroll
  for (int i = 0; i < 2; ++i)
#pragma unroll
    for (int j = 0; j < 2; ++j)
#pragma unroll
      for (int v = 0; v < 8; ++v) acc[i][j][v] = 0.0f;

  for (int kb = 0; kb < n; kb += 32) {
    // ---- A tile: 64 rows x 32 cols of bf16 ----
#if defined(HAVE_TDM)
    if (wave == 0) {
      // 2-D Tensor DMA descriptor (D#), groups per CDNA5 ISA.
      const unsigned long long ga =
          (unsigned long long)(A + (size_t)m0 * n + kb);   // tile start (bytes)
      const unsigned lds_a =
          (unsigned)(unsigned long long)(__attribute__((address_space(3))) char*)(&sA[0][0]);
      u32x4 g0;
      g0[0] = 1u;                                          // count=1, user D#
      g0[1] = lds_a;                                       // lds_addr
      g0[2] = (unsigned)ga;                                // global_addr[31:0]
      g0[3] = (unsigned)((ga >> 32) & 0x01ffffffull)       // global_addr[56:32]
              | (2u << 30);                                // type=2 ("image")
      i32x8 g1;
      g1[0] = 0x00010000;                                  // data_size=1 (2B)
      g1[1] = (int)((unsigned)n << 16);                    // tensor_dim0[15:0]
      g1[2] = (int)(((unsigned)n >> 16) | ((unsigned)n << 16)); // d0 hi | d1 lo
      g1[3] = (int)(((unsigned)n >> 16) | (32u << 16));    // d1 hi | tile_dim0=32
      g1[4] = 64;                                          // tile_dim1=64, tile_dim2=0
      g1[5] = n;                                           // dim0_stride[31:0]
      g1[6] = 0;                                           // dim0_stride hi, d1 stride lo
      g1[7] = 0;
      i32x4 gz;  gz[0] = 0; gz[1] = 0; gz[2] = 0; gz[3] = 0;
      i32x8 gz8;
#pragma unroll
      for (int z = 0; z < 8; ++z) gz8[z] = 0;
      __builtin_amdgcn_tensor_load_to_lds(g0, g1, gz, gz, gz8, 0);
      __builtin_amdgcn_s_wait_tensorcnt(0);
    }
#elif defined(HAVE_ASYNC_LDS)
    for (int q = tid; q < 256; q += 128) {                 // 64 rows x 4 chunks
      const int row = q >> 2;
      const int c8  = (q & 3) * 8;
      __builtin_amdgcn_global_load_async_to_lds_b128(
          GLOBAL_V4I(A + (size_t)(m0 + row) * n + kb + c8),
          LDS_V4I(&sA[row][c8]), 0, 0);
    }
#else
#pragma unroll
    for (int q = tid; q < 256; q += 128) {                 // 64 rows x 4 chunks
      const int row = q >> 2;
      const int c8  = (q & 3) * 8;
      *reinterpret_cast<uint4*>(&sA[row][c8]) =
          *reinterpret_cast<const uint4*>(A + (size_t)(m0 + row) * n + kb + c8);
    }
#endif

    // ---- B tile: 32 rows x 64 cols of bf16 ----
#if defined(HAVE_ASYNC_LDS)
    for (int q = tid; q < 256; q += 128) {                 // 32 rows x 8 chunks
      const int row = q >> 3;
      const int c8  = (q & 7) * 8;
      __builtin_amdgcn_global_load_async_to_lds_b128(
          GLOBAL_V4I(B + (size_t)(kb + row) * n + n0 + c8),
          LDS_V4I(&sB[row][c8]), 0, 0);
    }
    __builtin_amdgcn_s_wait_asynccnt(0);
#else
#pragma unroll
    for (int q = tid; q < 256; q += 128) {                 // 32 rows x 8 chunks
      const int row = q >> 3;
      const int c8  = (q & 7) * 8;
      *reinterpret_cast<uint4*>(&sB[row][c8]) =
          *reinterpret_cast<const uint4*>(B + (size_t)(kb + row) * n + n0 + c8);
    }
    if (kb + 32 < n) {                                     // prefetch next K tile
      __builtin_prefetch(A + (size_t)(m0 + (tid >> 1)) * n + kb + 32, 0, 3);
      __builtin_prefetch(B + (size_t)(kb + 32 + (tid >> 2)) * n + n0, 0, 3);
    }
#endif
    __syncthreads();

    // Pack fragments: contiguous k order (k = 16*hi + e), same for A and B.
    v16bf af[2], bf[2];
#pragma unroll
    for (int t = 0; t < 2; ++t) {
      const bf16_t* ap = &sA[wm + t * 16 + l16][hi * 16];
#pragma unroll
      for (int e = 0; e < 8; ++e) {          // lanes<16: K 0-15; hi: K 16-31
        af[t][e]     = ap[e];
        af[t][e + 8] = ap[8 + e];
      }
      const int bcol = wn + t * 16 + l16;
#pragma unroll
      for (int e = 0; e < 16; ++e)           // lanes<16: K 0-15; hi: K 16-31
        bf[t][e] = sB[hi * 16 + e][bcol];
    }
#pragma unroll
    for (int i = 0; i < 2; ++i)
#pragma unroll
      for (int j = 0; j < 2; ++j)
        acc[i][j] = __builtin_amdgcn_wmma_f32_16x16x32_bf16(
            false, af[i], false, bf[j], (short)0, acc[i][j], false, false);
    __syncthreads();
  }

  // Store: 32-bit C/D layout — lane<16: M = v, N = lane; lane>=16: M = 8+v.
#pragma unroll
  for (int i = 0; i < 2; ++i)
#pragma unroll
    for (int j = 0; j < 2; ++j) {
      const int rbase = m0 + wm + i * 16 + hi * 8;
      const int col   = n0 + wn + j * 16 + l16;
#pragma unroll
      for (int v = 0; v < 8; ++v)
        C[(size_t)(rbase + v) * n + col] = acc[i][j][v];
    }
}

// ---------------------------------------------------------------------------
// Adjacency construction / fixups
// ---------------------------------------------------------------------------
__global__ void scatter_edges(bf16_t* A, const int* __restrict__ ei, int E, int n) {
  int e = blockIdx.x * blockDim.x + threadIdx.x;
  if (e < E) {
    int r = ei[e], c = ei[E + e];
    A[(size_t)r * n + c] = (bf16_t)1.0f;
  }
}

__global__ void add_diag_bf16(bf16_t* A, int n) {
  int i = blockIdx.x * blockDim.x + threadIdx.x;
  if (i < n) {
    size_t o = (size_t)i * n + i;
    A[o] = (bf16_t)((float)A[o] + 1.0f);
  }
}

__global__ void zero_diag_f32(float* C, int n) {
  int i = blockIdx.x * blockDim.x + threadIdx.x;
  if (i < n) C[(size_t)i * n + i] = 0.0f;
}

// ---------------------------------------------------------------------------
// TopK pooling: score, bitonic sort, gathers
// ---------------------------------------------------------------------------
__global__ __launch_bounds__(128) void score_kernel(const float* __restrict__ X,
                                                    const float* __restrict__ w,
                                                    float* __restrict__ score) {
  __shared__ float rd[CH];
  __shared__ float rw[CH];
  const int r = blockIdx.x, t = threadIdx.x;
  float wv = w[t];
  rd[t] = X[(size_t)r * CH + t] * wv;
  rw[t] = wv * wv;
  __syncthreads();
  for (int s = 64; s > 0; s >>= 1) {
    if (t < s) { rd[t] += rd[t + s]; rw[t] += rw[t + s]; }
    __syncthreads();
  }
  if (t == 0) score[r] = tanhf(rd[0] * rsqrtf(rw[0]));
}

// Single-workgroup bitonic sort, descending value, ties -> lower index first.
__global__ __launch_bounds__(1024) void topk_sort(const float* __restrict__ score,
                                                  int n, int k,
                                                  int* __restrict__ perm,
                                                  float* __restrict__ vals) {
  __shared__ float sv[4096];
  __shared__ int   si[4096];
  const int t = threadIdx.x;
  for (int i = t; i < n; i += 1024) { sv[i] = score[i]; si[i] = i; }
  __syncthreads();
  for (int ksz = 2; ksz <= n; ksz <<= 1) {
    for (int j = ksz >> 1; j > 0; j >>= 1) {
      for (int i = t; i < n; i += 1024) {
        int ixj = i ^ j;
        if (ixj > i) {
          float va = sv[i], vb = sv[ixj];
          int   ia = si[i], ib = si[ixj];
          bool g    = (va > vb) || (va == vb && ia < ib); // i ranks before ixj
          bool desc = ((i & ksz) == 0);
          if (desc ? !g : g) {
            sv[i] = vb; sv[ixj] = va; si[i] = ib; si[ixj] = ia;
          }
        }
      }
      __syncthreads();
    }
  }
  for (int i = t; i < k; i += 1024) { perm[i] = si[i]; vals[i] = sv[i]; }
}

__global__ void pool_gather_x(const float* __restrict__ xin,
                              const int* __restrict__ perm,
                              const float* __restrict__ vals,
                              float* __restrict__ xout) {
  const int i = blockIdx.x, ch = threadIdx.x;
  xout[(size_t)i * CH + ch] = xin[(size_t)perm[i] * CH + ch] * vals[i];
}

// A_new = A_aug[perm][:,perm] (f32) and bf16(A_new + I) for the next augment.
__global__ void pool_gather_A(const float* __restrict__ Aaug,
                              const int* __restrict__ perm, int n, int k,
                              float* __restrict__ Anew,
                              bf16_t* __restrict__ Abf) {
  size_t gid = (size_t)blockIdx.x * blockDim.x + threadIdx.x;
  if (gid >= (size_t)k * k) return;
  int i = (int)(gid / k), j = (int)(gid % k);
  float v = Aaug[(size_t)perm[i] * n + perm[j]];
  Anew[gid] = v;
  Abf[gid]  = (bf16_t)(v + (i == j ? 1.0f : 0.0f));
}

// ---------------------------------------------------------------------------
// GCN (improved): d = rsqrt(colsum(A) + 2); Z = d.*(xW); out = relu(d.*(A^T Z + 2Z) + b)
// ---------------------------------------------------------------------------
__global__ void colsum_dinv(const float* __restrict__ A, float* __restrict__ dinv, int k) {
  int c = blockIdx.x * blockDim.x + threadIdx.x;
  if (c >= k) return;
  float s = 2.0f;
  for (int r = 0; r < k; ++r) s += A[(size_t)r * k + c];
  dinv[c] = rsqrtf(s);
}

__global__ __launch_bounds__(128) void xw_scale(const float* __restrict__ X,
                                                const float* __restrict__ W,
                                                const float* __restrict__ dinv,
                                                float* __restrict__ Z) {
  __shared__ float xr[CH];
  const int r = blockIdx.x, ch = threadIdx.x;
  xr[ch] = X[(size_t)r * CH + ch];
  __syncthreads();
  float acc = 0.0f;
#pragma unroll 8
  for (int t = 0; t < CH; ++t) acc += xr[t] * W[t * CH + ch];
  Z[(size_t)r * CH + ch] = dinv[r] * acc;
}

__global__ __launch_bounds__(128) void gcn_aggregate(const float* __restrict__ A,
                                                     const float* __restrict__ Z,
                                                     const float* __restrict__ dinv,
                                                     const float* __restrict__ bias,
                                                     float* __restrict__ out, int k) {
  __shared__ float acol[128];
  const int c = blockIdx.x, ch = threadIdx.x;
  float acc = 2.0f * Z[(size_t)c * CH + ch];
  for (int r0 = 0; r0 < k; r0 += 128) {
    __syncthreads();
    acol[ch] = A[(size_t)(r0 + ch) * k + c];
    __syncthreads();
#pragma unroll 8
    for (int rr = 0; rr < 128; ++rr)
      acc += acol[rr] * Z[(size_t)(r0 + rr) * CH + ch];
  }
  out[(size_t)c * CH + ch] = fmaxf(dinv[c] * acc + bias[ch], 0.0f);
}

// ---------------------------------------------------------------------------
// Unpool: U pre-filled with xs[j]; overwrite rows perm[i] with xs[j] + x_up.
// perm entries are distinct -> no atomics needed.
// ---------------------------------------------------------------------------
__global__ void unpool_scatter(const float* __restrict__ xsj,
                               const float* __restrict__ xup,
                               const int* __restrict__ perm,
                               float* __restrict__ U) {
  const int i = blockIdx.x, ch = threadIdx.x;
  const int p = perm[i];
  U[(size_t)p * CH + ch] = xsj[(size_t)p * CH + ch] + xup[(size_t)i * CH + ch];
}

// ---------------------------------------------------------------------------
// Host orchestration
// ---------------------------------------------------------------------------
extern "C" void kernel_launch(void* const* d_in, const int* in_sizes, int n_in,
                              void* d_out, int out_size, void* d_ws, size_t ws_size,
                              hipStream_t stream) {
  (void)in_sizes; (void)n_in; (void)out_size; (void)ws_size;
  const int N0 = 4096, K0 = 2048, K1 = 1024, K2 = 512, E = 65536;

  const float* x_in   = (const float*)d_in[0];
  const int*   eidx   = (const int*)d_in[1];
  const float* pool_w = (const float*)d_in[2];
  const float* down_W = (const float*)d_in[3];
  const float* down_b = (const float*)d_in[4];
  const float* up_W   = (const float*)d_in[5];
  const float* up_b   = (const float*)d_in[6];
  float*       out    = (float*)d_out;

  char* ws = (char*)d_ws;
  size_t off = 0;
  auto alloc = [&](size_t bytes) -> void* {
    off = (off + 255) & ~(size_t)255;
    void* p = ws + off;
    off += bytes;
    return p;
  };

  bf16_t* bfA  = (bf16_t*)alloc((size_t)N0 * N0 * 2);  // reused every level
  float*  Caug = (float*) alloc((size_t)N0 * N0 * 4);  // reused every level
  float*  A1   = (float*) alloc((size_t)K0 * K0 * 4);  // As[1]
  float*  A2   = (float*) alloc((size_t)K1 * K1 * 4);  // As[2]
  float*  A3   = (float*) alloc((size_t)K2 * K2 * 4);  // depth-3 adjacency
  float*  x1   = (float*) alloc((size_t)K0 * CH * 4);
  float*  xd0  = (float*) alloc((size_t)K0 * CH * 4);  // xs[1]
  float*  x2   = (float*) alloc((size_t)K1 * CH * 4);
  float*  xd1  = (float*) alloc((size_t)K1 * CH * 4);  // xs[2]
  float*  x3   = (float*) alloc((size_t)K2 * CH * 4);
  float*  xd2  = (float*) alloc((size_t)K2 * CH * 4);
  float*  U2   = (float*) alloc((size_t)K1 * CH * 4);
  float*  xu2  = (float*) alloc((size_t)K1 * CH * 4);
  float*  U1   = (float*) alloc((size_t)K0 * CH * 4);
  float*  xu1  = (float*) alloc((size_t)K0 * CH * 4);
  float*  Z    = (float*) alloc((size_t)K0 * CH * 4);
  float*  dinv = (float*) alloc((size_t)K0 * 4);
  float*  scr  = (float*) alloc((size_t)N0 * 4);
  float*  vals = (float*) alloc((size_t)K0 * 4);
  int*    p0   = (int*)   alloc((size_t)K0 * 4);
  int*    p1   = (int*)   alloc((size_t)K1 * 4);
  int*    p2   = (int*)   alloc((size_t)K2 * 4);

  auto gcn = [&](const float* A, const float* X, const float* W, const float* b,
                 float* o, int k) {
    colsum_dinv<<<dim3((k + 255) / 256), dim3(256), 0, stream>>>(A, dinv, k);
    xw_scale<<<dim3(k), dim3(128), 0, stream>>>(X, W, dinv, Z);
    gcn_aggregate<<<dim3(k), dim3(128), 0, stream>>>(A, Z, dinv, b, o, k);
  };

  // ---- build bf16 (A + I) at n=4096 ----
  (void)hipMemsetAsync(bfA, 0, (size_t)N0 * N0 * 2, stream);
  scatter_edges<<<dim3(E / 256), dim3(256), 0, stream>>>(bfA, eidx, E, N0);
  add_diag_bf16<<<dim3(N0 / 256), dim3(256), 0, stream>>>(bfA, N0);

  // ---- down level 0 ----
  gemm_bf16_sq<<<dim3(N0 / 64, N0 / 64), dim3(128), 0, stream>>>(bfA, bfA, Caug, N0);
  zero_diag_f32<<<dim3(N0 / 256), dim3(256), 0, stream>>>(Caug, N0);
  score_kernel<<<dim3(N0), dim3(128), 0, stream>>>(x_in, pool_w + 0 * CH, scr);
  topk_sort<<<dim3(1), dim3(1024), 0, stream>>>(scr, N0, K0, p0, vals);
  pool_gather_x<<<dim3(K0), dim3(128), 0, stream>>>(x_in, p0, vals, x1);
  pool_gather_A<<<dim3((unsigned)(((size_t)K0 * K0 + 255) / 256)), dim3(256), 0, stream>>>(
      Caug, p0, N0, K0, A1, bfA);
  gcn(A1, x1, down_W + 0 * CH * CH, down_b + 0 * CH, xd0, K0);

  // ---- down level 1 ----
  gemm_bf16_sq<<<dim3(K0 / 64, K0 / 64), dim3(128), 0, stream>>>(bfA, bfA, Caug, K0);
  zero_diag_f32<<<dim3(K0 / 256), dim3(256), 0, stream>>>(Caug, K0);
  score_kernel<<<dim3(K0), dim3(128), 0, stream>>>(xd0, pool_w + 1 * CH, scr);
  topk_sort<<<dim3(1), dim3(1024), 0, stream>>>(scr, K0, K1, p1, vals);
  pool_gather_x<<<dim3(K1), dim3(128), 0, stream>>>(xd0, p1, vals, x2);
  pool_gather_A<<<dim3((unsigned)(((size_t)K1 * K1 + 255) / 256)), dim3(256), 0, stream>>>(
      Caug, p1, K0, K1, A2, bfA);
  gcn(A2, x2, down_W + 1 * CH * CH, down_b + 1 * CH, xd1, K1);

  // ---- down level 2 ----
  gemm_bf16_sq<<<dim3(K1 / 64, K1 / 64), dim3(128), 0, stream>>>(bfA, bfA, Caug, K1);
  zero_diag_f32<<<dim3(K1 / 256), dim3(256), 0, stream>>>(Caug, K1);
  score_kernel<<<dim3(K1), dim3(128), 0, stream>>>(xd1, pool_w + 2 * CH, scr);
  topk_sort<<<dim3(1), dim3(1024), 0, stream>>>(scr, K1, K2, p2, vals);
  pool_gather_x<<<dim3(K2), dim3(128), 0, stream>>>(xd1, p2, vals, x3);
  pool_gather_A<<<dim3((unsigned)(((size_t)K2 * K2 + 255) / 256)), dim3(256), 0, stream>>>(
      Caug, p2, K1, K2, A3, bfA);
  gcn(A3, x3, down_W + 2 * CH * CH, down_b + 2 * CH, xd2, K2);

  // ---- up i=0 (j=2) ----
  (void)hipMemcpyAsync(U2, xd1, (size_t)K1 * CH * 4, hipMemcpyDeviceToDevice, stream);
  unpool_scatter<<<dim3(K2), dim3(128), 0, stream>>>(xd1, xd2, p2, U2);
  gcn(A2, U2, up_W + 0 * CH * CH, up_b + 0 * CH, xu2, K1);

  // ---- up i=1 (j=1) ----
  (void)hipMemcpyAsync(U1, xd0, (size_t)K0 * CH * 4, hipMemcpyDeviceToDevice, stream);
  unpool_scatter<<<dim3(K1), dim3(128), 0, stream>>>(xd0, xu2, p1, U1);
  gcn(A1, U1, up_W + 1 * CH * CH, up_b + 1 * CH, xu1, K0);

  // ---- up i=2 (j=0): out = x_in + scatter(p0, xu1), no GCN ----
  (void)hipMemcpyAsync(out, x_in, (size_t)N0 * CH * 4, hipMemcpyDeviceToDevice, stream);
  unpool_scatter<<<dim3(K0), dim3(128), 0, stream>>>(x_in, xu1, p0, out);
}
